// SFM_LSTM_24034636988678
// MI455X (gfx1250) — matrix operand order repaired and checked by reading, not compiled, not run on hardware
//
#include <hip/hip_runtime.h>
#include <hip/hip_bf16.h>
#include <math.h>

typedef __attribute__((ext_vector_type(16))) _Float16 v16h;
typedef __attribute__((ext_vector_type(8)))  _Float16 v8h;
typedef __attribute__((ext_vector_type(8)))  float    v8f;
typedef __attribute__((ext_vector_type(4)))  float    fx4;

#define B_TOT 1024
#define Lc 32
#define Cc 64
#define Hc 128
#define Fc 32
#define ACT_STRIDE 324   // fp32 rows: 324 % 64 == 4 -> conflict-free
#define AH_STRIDE  328   // f16 rows: 656B = 164 dwords, 164 % 64 == 36 -> conflict-free, 16B aligned

// ---- swizzled f16 weight region (offsets in halves) ----
// tile = 512 halves, order per weight: (nt * KC + kc)
#define OQ    0        // Q    (128x64):  KC=4,  NT=4  -> 16 tiles
#define OR_   8192     // R    (64x128):  KC=2,  NT=8  -> 16 tiles
#define OWI   16384    // Wi   (192x128): KC=6,  NT=8  -> 48 tiles
#define OWG   40960    // Wg
#define OWSTE 65536    // Wste
#define OWFRE 90112    // Wfre (192x32):  KC=6,  NT=2  -> 12 tiles
#define OWOM  96256    // Wom
#define OWO   102400   // Wo   (320x128): KC=10, NT=8  -> 80 tiles
#define SWZ_HALVES 143360

// ---- fp32 workspace region (offsets in floats from ws base) ----
#define XT_OFF   71680      // B*64   mogrified x_t
#define HMOG_OFF 137216     // B*128  mogrified h
#define IT_OFF   268288     // B*128  i_t
#define GT_OFF   399360     // B*128  c_hat
#define FSTE_OFF 530432     // B*128  f_ste
#define FFRE_OFF 661504     // B*32   f_fre
#define OM_OFF   694272     // B*32   omega
#define CT_OFF   727040     // B*128  c_t
#define RE_OFF   858112     // B*128*32
#define IM_OFF   5052416    // B*128*32

#define WMMA_F16(a,b,c) __builtin_amdgcn_wmma_f32_16x16x32_f16(false,(a),false,(b),(short)0,(c),false,false)

__device__ __forceinline__ float sigf(float x) { return 1.0f / (1.0f + __expf(-x)); }

// A-fragment (16x32 f16) straight from the f16 LDS mirror: 2x ds_load_b128, no cvt.
// ISA layout: lane l -> row m = l&15; half i -> k = 8*(l>>4) + 16*(i>>3) + (i&7)
__device__ __forceinline__ v16h load_A16(const _Float16* acth, int lane, int k0) {
    int m = lane & 15;
    int base = (lane >> 4) << 3;
    const _Float16* p = acth + m * AH_STRIDE + k0 + base;
    v8h lo = *(const v8h*)(p);
    v8h hi = *(const v8h*)(p + 16);
    return __builtin_shufflevector(lo, hi, 0,1,2,3,4,5,6,7,8,9,10,11,12,13,14,15);
}

// B-fragment: pre-swizzled, lane reads 16 contiguous halves (32B, coalesced)
__device__ __forceinline__ v16h load_B(const _Float16* wswz, int tile, int lane) {
    const v16h* p = (const v16h*)(wswz + (size_t)tile * 512);
    return p[lane];
}

// dual-buffer activation store: fp32 master + f16 mirror (cvt paid once per write)
__device__ __forceinline__ void store2(float* actf, _Float16* acth, int m, int col, float v) {
    actf[m * ACT_STRIDE + col] = v;
    acth[m * AH_STRIDE + col]  = (_Float16)v;
}

// ---- one-shot weight f32->f16 swizzle into WMMA B-fragment order ----
// B layout: lane l -> col n = l&15; half i -> k = 8*(l>>4) + 16*(i>>3) + (i&7)
__global__ __launch_bounds__(512) void kswz(
    const float* __restrict__ Q, const float* __restrict__ R,
    const float* __restrict__ Wi, const float* __restrict__ Wg,
    const float* __restrict__ Wste, const float* __restrict__ Wfre,
    const float* __restrict__ Wom, const float* __restrict__ Wo,
    _Float16* __restrict__ swz)
{
    int tile = blockIdx.x; // 0..279
    const float* W; int N, KC, tl; long off;
    if      (tile < 16)  { W = Q;    N = 64;  KC = 4;  tl = tile;       off = OQ;    }
    else if (tile < 32)  { W = R;    N = 128; KC = 2;  tl = tile - 16;  off = OR_;   }
    else if (tile < 80)  { W = Wi;   N = 128; KC = 6;  tl = tile - 32;  off = OWI;   }
    else if (tile < 128) { W = Wg;   N = 128; KC = 6;  tl = tile - 80;  off = OWG;   }
    else if (tile < 176) { W = Wste; N = 128; KC = 6;  tl = tile - 128; off = OWSTE; }
    else if (tile < 188) { W = Wfre; N = 32;  KC = 6;  tl = tile - 176; off = OWFRE; }
    else if (tile < 200) { W = Wom;  N = 32;  KC = 6;  tl = tile - 188; off = OWOM;  }
    else                 { W = Wo;   N = 128; KC = 10; tl = tile - 200; off = OWO;   }
    int kc = tl % KC, nt = tl / KC;
    int tid = threadIdx.x;
    int lane = tid >> 4, i = tid & 15;
    int n = nt * 16 + (lane & 15);
    int k = kc * 32 + ((lane >> 4) << 3) + ((i >> 3) << 4) + (i & 7);
    swz[off + (long)tl * 512 + tid] = (_Float16)W[k * N + n];
}

// ---- per-step gate kernel: prev output gate + mogrify + gate GEMMs ----
// 16 batch rows per block, 8 wave32s
__global__ __launch_bounds__(256) void kgates(
    const float* __restrict__ x, const _Float16* __restrict__ swz,
    float* __restrict__ wsf,
    const float* __restrict__ bi, const float* __restrict__ bg,
    const float* __restrict__ bste, const float* __restrict__ bfre,
    const float* __restrict__ bom, const float* __restrict__ bo, int t)
{
    // cols 0..63 xt | 64..191 h | 192..319 c_prev
    __shared__ float    actf[16 * ACT_STRIDE];
    __shared__ _Float16 acth[16 * AH_STRIDE];
    const int tid = threadIdx.x, lane = tid & 31, wv = tid >> 5;
    const int b0 = blockIdx.x * 16;
    float* xt_ws = wsf + XT_OFF;
    float* hm_ws = wsf + HMOG_OFF;
    float* ct_ws = wsf + CT_OFF;

    // ---- phase 1: h_in = sig(oh @ Wo + bo) * tanh(c_prev), or 0 at t==0
    if (t == 0) {
        for (int idx = tid; idx < 16 * Hc; idx += 256) {
            int m = idx >> 7, n = idx & 127;
            store2(actf, acth, m, 64 + n, 0.f);
        }
    } else {
        for (int idx = tid; idx < 16 * Cc; idx += 256) {
            int m = idx >> 6, n = idx & 63;
            store2(actf, acth, m, n, xt_ws[(b0 + m) * Cc + n]);
        }
        for (int idx = tid; idx < 16 * Hc; idx += 256) {
            int m = idx >> 7, n = idx & 127;
            store2(actf, acth, m, 64 + n,  hm_ws[(b0 + m) * Hc + n]);
            store2(actf, acth, m, 192 + n, ct_ws[(b0 + m) * Hc + n]);
        }
        __syncthreads();
        v8f acc = {};
#pragma unroll
        for (int kc = 0; kc < 10; ++kc) {
            v16h a  = load_A16(&acth[0], lane, kc * 32);
            v16h bw = load_B(swz + OWO, wv * 10 + kc, lane);
            acc = WMMA_F16(a, bw, acc);
        }
        int nl = lane & 15, mb = (lane >> 4) << 3;
        int n = wv * 16 + nl;
        float bias = bo[n];
        float hnew[8];
#pragma unroll
        for (int r = 0; r < 8; ++r) {
            float c = actf[(mb + r) * ACT_STRIDE + 192 + n];
            hnew[r] = sigf(acc[r] + bias) * tanhf(c);
        }
        __syncthreads();  // all A reads of h/c done before overwriting h
#pragma unroll
        for (int r = 0; r < 8; ++r)
            store2(actf, acth, mb + r, 64 + n, hnew[r]);
    }
    __syncthreads();

    // load current x_t into cols 0..63
    for (int idx = tid; idx < 16 * Cc; idx += 256) {
        int m = idx >> 6, n = idx & 63;
        store2(actf, acth, m, n, x[((size_t)(b0 + m) * Lc + t) * Cc + n]);
    }
    __syncthreads();

    // ---- phase 2: mogrify, 5 rounds (even: xt update via h@Q; odd: h update via xt@R)
#pragma unroll
    for (int rnd = 0; rnd < 5; ++rnd) {
        const bool even = ((rnd & 1) == 0);
        v8f acc = {};
        if (even) {
            if (wv < 4) {
#pragma unroll
                for (int kc = 0; kc < 4; ++kc) {
                    v16h a  = load_A16(&acth[0], lane, 64 + kc * 32);
                    v16h bw = load_B(swz + OQ, wv * 4 + kc, lane);
                    acc = WMMA_F16(a, bw, acc);
                }
            }
        } else {
#pragma unroll
            for (int kc = 0; kc < 2; ++kc) {
                v16h a  = load_A16(&acth[0], lane, kc * 32);
                v16h bw = load_B(swz + OR_, wv * 2 + kc, lane);
                acc = WMMA_F16(a, bw, acc);
            }
        }
        __syncthreads();
        int nl = lane & 15, mb = (lane >> 4) << 3;
        if (even) {
            if (wv < 4) {
                int n = wv * 16 + nl;
#pragma unroll
                for (int r = 0; r < 8; ++r) {
                    float v = actf[(mb + r) * ACT_STRIDE + n];
                    store2(actf, acth, mb + r, n, 2.f * sigf(acc[r]) * v);
                }
            }
        } else {
            int n = wv * 16 + nl;
#pragma unroll
            for (int r = 0; r < 8; ++r) {
                float v = actf[(mb + r) * ACT_STRIDE + 64 + n];
                store2(actf, acth, mb + r, 64 + n, 2.f * sigf(acc[r]) * v);
            }
        }
        __syncthreads();
    }

    // ---- phase 3: gate GEMMs on ih = [xt | h] (K=192), 28 output tiles
    for (int T = wv; T < 28; T += 8) {
        const _Float16* wb; int nt;
        if      (T < 8)  { wb = swz + OWI;   nt = T;      }
        else if (T < 16) { wb = swz + OWG;   nt = T - 8;  }
        else if (T < 24) { wb = swz + OWSTE; nt = T - 16; }
        else if (T < 26) { wb = swz + OWFRE; nt = T - 24; }
        else             { wb = swz + OWOM;  nt = T - 26; }
        v8f acc = {};
#pragma unroll
        for (int kc = 0; kc < 6; ++kc) {
            v16h a  = load_A16(&acth[0], lane, kc * 32);
            v16h bw = load_B(wb, nt * 6 + kc, lane);
            acc = WMMA_F16(a, bw, acc);
        }
        int nl = lane & 15, mb = (lane >> 4) << 3;
        int n = nt * 16 + nl;
        float bias; float* dst; float scale = 1.f; int stride;
        if      (T < 8)  { bias = bi[n];   dst = wsf + IT_OFF;   stride = Hc; }
        else if (T < 16) { bias = bg[n];   dst = wsf + GT_OFF;   stride = Hc; }
        else if (T < 24) { bias = bste[n]; dst = wsf + FSTE_OFF; stride = Hc; }
        else if (T < 26) { bias = bfre[n]; dst = wsf + FFRE_OFF; stride = Fc; }
        else             { bias = bom[n];  dst = wsf + OM_OFF;   stride = Fc; scale = 6.283185307179586f; }
#pragma unroll
        for (int r = 0; r < 8; ++r) {
            int m = mb + r;
            dst[(b0 + m) * stride + n] = scale * sigf(acc[r] + bias);
        }
    }

    // persist mogrified xt and h for next step / final kernel
    for (int idx = tid; idx < 16 * Cc; idx += 256) {
        int m = idx >> 6, n = idx & 63;
        xt_ws[(b0 + m) * Cc + n] = actf[m * ACT_STRIDE + n];
    }
    for (int idx = tid; idx < 16 * Hc; idx += 256) {
        int m = idx >> 7, n = idx & 127;
        hm_ws[(b0 + m) * Hc + n] = actf[m * ACT_STRIDE + 64 + n];
    }
}

// ---- per-step frequency-state update: one block per batch row, thread = h ----
__global__ __launch_bounds__(128) void kstate(
    float* __restrict__ wsf, const float* __restrict__ Wa,
    const float* __restrict__ ba, int t)
{
    __shared__ float sco[Fc], ssi[Fc], sff[Fc], swa[Fc];
    const int b = blockIdx.x, h = threadIdx.x;
    if (h < Fc) {
        float om = wsf[OM_OFF + b * Fc + h];
        sco[h] = cosf(om);
        ssi[h] = sinf(om);
        sff[h] = wsf[FFRE_OFF + b * Fc + h];
        swa[h] = Wa[h];
    }
    __syncthreads();
    float it = wsf[IT_OFF + b * Hc + h];
    float gt = wsf[GT_OFF + b * Hc + h];
    float fs = wsf[FSTE_OFF + b * Hc + h];
    float ic = it * gt;
    float* re = wsf + RE_OFF + (size_t)(b * Hc + h) * Fc;
    float* im = wsf + IM_OFF + (size_t)(b * Hc + h) * Fc;
    float acc = 0.f;
#pragma unroll
    for (int f0 = 0; f0 < Fc; f0 += 4) {
        fx4 r = {}, q = {};
        if (t != 0) { r = *(const fx4*)(re + f0); q = *(const fx4*)(im + f0); }
#pragma unroll
        for (int j = 0; j < 4; ++j) {
            int f = f0 + j;
            float ft = fs * sff[f];
            float rn = ft * r[j] + ic * sco[f];
            float qn = ft * q[j] + ic * ssi[f];
            r[j] = rn; q[j] = qn;
            acc += sqrtf(rn * rn + qn * qn) * swa[f];
        }
        *(fx4*)(re + f0) = r;
        *(fx4*)(im + f0) = q;
    }
    wsf[CT_OFF + b * Hc + h] = tanhf(acc + ba[0]);
}

// ---- final: last output gate + h @ Wout + bout ----
__global__ __launch_bounds__(256) void kfinal(
    const _Float16* __restrict__ swz, const float* __restrict__ wsf,
    const float* __restrict__ bo, const float* __restrict__ Wout,
    const float* __restrict__ bout, float* __restrict__ out)
{
    __shared__ float    actf[16 * ACT_STRIDE];
    __shared__ _Float16 acth[16 * AH_STRIDE];
    __shared__ float hl[16 * Hc];
    const int tid = threadIdx.x, lane = tid & 31, wv = tid >> 5;
    const int b0 = blockIdx.x * 16;
    const float* xt_ws = wsf + XT_OFF;
    const float* hm_ws = wsf + HMOG_OFF;
    const float* ct_ws = wsf + CT_OFF;
    for (int idx = tid; idx < 16 * Cc; idx += 256) {
        int m = idx >> 6, n = idx & 63;
        store2(actf, acth, m, n, xt_ws[(b0 + m) * Cc + n]);
    }
    for (int idx = tid; idx < 16 * Hc; idx += 256) {
        int m = idx >> 7, n = idx & 127;
        store2(actf, acth, m, 64 + n,  hm_ws[(b0 + m) * Hc + n]);
        store2(actf, acth, m, 192 + n, ct_ws[(b0 + m) * Hc + n]);
    }
    __syncthreads();
    v8f acc = {};
#pragma unroll
    for (int kc = 0; kc < 10; ++kc) {
        v16h a  = load_A16(&acth[0], lane, kc * 32);
        v16h bw = load_B(swz + OWO, wv * 10 + kc, lane);
        acc = WMMA_F16(a, bw, acc);
    }
    int nl = lane & 15, mb = (lane >> 4) << 3;
    int n = wv * 16 + nl;
    float bias = bo[n];
#pragma unroll
    for (int r = 0; r < 8; ++r) {
        int m = mb + r;
        float c = actf[m * ACT_STRIDE + 192 + n];
        hl[m * Hc + n] = sigf(acc[r] + bias) * tanhf(c);
    }
    __syncthreads();
    if (tid < 16) {
        float s = bout[0];
        const float* hr = &hl[tid * Hc];
        for (int k = 0; k < Hc; ++k) s += hr[k] * Wout[k];
        out[b0 + tid] = s;
    }
}

extern "C" void kernel_launch(void* const* d_in, const int* in_sizes, int n_in,
                              void* d_out, int out_size, void* d_ws, size_t ws_size,
                              hipStream_t stream) {
    (void)in_sizes; (void)n_in; (void)out_size; (void)ws_size;
    const float* x    = (const float*)d_in[0];
    const float* Q    = (const float*)d_in[1];
    const float* R    = (const float*)d_in[2];
    const float* Wi   = (const float*)d_in[3];
    const float* bi   = (const float*)d_in[4];
    const float* Wg   = (const float*)d_in[5];
    const float* bg   = (const float*)d_in[6];
    const float* Wste = (const float*)d_in[7];
    const float* bste = (const float*)d_in[8];
    const float* Wfre = (const float*)d_in[9];
    const float* bfre = (const float*)d_in[10];
    const float* Wom  = (const float*)d_in[11];
    const float* bom  = (const float*)d_in[12];
    const float* Wo   = (const float*)d_in[13];
    const float* bo   = (const float*)d_in[14];
    const float* Wa   = (const float*)d_in[15];
    const float* ba   = (const float*)d_in[16];
    const float* Wout = (const float*)d_in[17];
    const float* bout = (const float*)d_in[18];

    _Float16* swz = (_Float16*)d_ws;
    float* wsf = (float*)d_ws;
    float* out = (float*)d_out;

    kswz<<<280, 512, 0, stream>>>(Q, R, Wi, Wg, Wste, Wfre, Wom, Wo, swz);
    for (int t = 0; t < Lc; ++t) {
        kgates<<<B_TOT / 16, 256, 0, stream>>>(x, swz, wsf, bi, bg, bste, bfre, bom, bo, t);
        kstate<<<B_TOT, 128, 0, stream>>>(wsf, Wa, ba, t);
    }
    kfinal<<<B_TOT / 16, 256, 0, stream>>>(swz, wsf, bo, Wout, bout, out);
}
